// Bumblebee_16252156248475
// MI455X (gfx1250) — matrix operand order repaired and checked by reading, not compiled, not run on hardware
//
#include <hip/hip_runtime.h>

// ---------------------------------------------------------------------------
// Problem constants (fixed by the reference): B=8, T=2048, D_IN=1024, NODES=1024
// ---------------------------------------------------------------------------
constexpr int Bb = 8;
constexpr int Tt = 2048;
constexpr int Dd = 1024;
constexpr int Nd = 1024;

typedef __attribute__((ext_vector_type(16))) __bf16 v16bf;
typedef __attribute__((ext_vector_type(8)))  __bf16 bfx8;
typedef __attribute__((ext_vector_type(8)))  float  v8f;
typedef __attribute__((ext_vector_type(4)))  float  f32x4;

union FragAB { v16bf v; bfx8 h[2]; };

// CDNA5 async global->LDS DMA (tracked by ASYNCcnt). vdst = per-lane LDS byte
// offset (32-bit), vaddr = per-lane 64-bit global address; 16B per lane.
#define ASYNC_B128(ldsoff, gptr)                                   \
  asm volatile("global_load_async_to_lds_b128 %0, %1, off"         \
               :: "v"(ldsoff), "v"(gptr) : "memory")

__device__ __forceinline__ void wait_async0() {
#if __has_builtin(__builtin_amdgcn_s_wait_asynccnt)
  __builtin_amdgcn_s_wait_asynccnt(0);
#else
  asm volatile("s_wait_asynccnt 0x0" ::: "memory");
#endif
}

__device__ __forceinline__ unsigned lds_off(const void* p) {
  // AS3->flat keeps the group offset in the low 32 bits on gfx12.
  return (unsigned)(unsigned long long)p;
}

// ---------------------------------------------------------------------------
// Convert f32 -> bf16, same layout (coalesced both sides)
// ---------------------------------------------------------------------------
__global__ __launch_bounds__(256) void k_cvt_bf16(const float* __restrict__ x,
                                                  __bf16* __restrict__ y) {
  long i = ((long)blockIdx.x * blockDim.x + threadIdx.x) * 4;
  f32x4 v = *(const f32x4*)(x + i);
  y[i + 0] = (__bf16)v[0];
  y[i + 1] = (__bf16)v[1];
  y[i + 2] = (__bf16)v[2];
  y[i + 3] = (__bf16)v[3];
}

// ---------------------------------------------------------------------------
// Transpose-convert W [D][N] f32 -> Wt [N][D] bf16, via 32x32 LDS tile
// ---------------------------------------------------------------------------
__global__ __launch_bounds__(256) void k_cvt_wt(const float* __restrict__ w,
                                                __bf16* __restrict__ wt) {
  __shared__ float tile[32][33];
  const int n0 = blockIdx.x * 32;
  const int d0 = blockIdx.y * 32;
  const int c = threadIdx.x & 31;
  const int r4 = (threadIdx.x >> 5) * 4;
#pragma unroll
  for (int i = 0; i < 4; ++i) {
    int r = r4 + i;
    tile[c][r] = w[(long)(d0 + r) * Nd + (n0 + c)];
  }
  __syncthreads();
#pragma unroll
  for (int i = 0; i < 4; ++i) {
    int r = r4 + i;
    wt[(long)(n0 + r) * Dd + (d0 + c)] = (__bf16)tile[r][c];
  }
}

// ---------------------------------------------------------------------------
// Generic bf16 WMMA GEMM:  C[m][n] = sum_k A[m][k] * Bt[n][k]
//   A  : [M][lda] bf16 row-major (batch stride sA elements)
//   Bt : [N][ldb] bf16 row-major, K-contiguous  (batch stride sB)
//   C  : f32 or bf16; optionally transposed store C[n*ldc + m]
// Block tile 128(M) x 256(N), BK=32, 256 threads = 8 waves (2 in M x 4 in N),
// wave tile 64x64 -> 4x4 v_wmma_f32_16x16x32_bf16 per K-step (16 WMMA : 16
// ds_load_b128). Tiles staged with GLOBAL_LOAD_ASYNC_TO_LDS_B128 (ASYNCcnt),
// double-buffered, one s_wait_asynccnt + barrier per K-step.
// ---------------------------------------------------------------------------
template <bool OUT_BF16, bool OUT_TRANS>
__global__ __launch_bounds__(256) void k_gemm(
    const __bf16* __restrict__ A, const __bf16* __restrict__ Bt,
    void* __restrict__ Cv, int K, int lda, int ldb, int ldc,
    long sA, long sB, long sC) {
  constexpr int BM = 128, BN = 256, BK = 32, LDW = BK + 8;
  __shared__ __bf16 As[2][BM * LDW];
  __shared__ __bf16 Bs[2][BN * LDW];

  const int bz = blockIdx.z;
  A  += (long)bz * sA;
  Bt += (long)bz * sB;

  const int tid  = threadIdx.x;
  const int lane = tid & 31;
  const int wave = tid >> 5;
  const int wm   = wave & 1;    // wave M offset = wm*64
  const int wn   = wave >> 1;   // wave N offset = wn*64
  const int hf   = lane >> 4;   // lane half (0/1)
  const int l16  = lane & 15;

  const long m0 = (long)blockIdx.y * BM;
  const long n0 = (long)blockIdx.x * BN;

  // Per-thread 8-bf16 (16B) chunks within the K-step tiles.
  const int e0  = tid * 8;
  const int ar0 = e0 >> 5;          // rows 0..63 (A covers +0,+64; B +0..+192)
  const int ac0 = e0 & 31;

  const __bf16* gA0 = A + (m0 + ar0) * lda + ac0;
  const __bf16* gA1 = gA0 + (long)64 * lda;
  const __bf16* gB0 = Bt + (n0 + ar0) * ldb + ac0;
  const __bf16* gB1 = gB0 + (long)64 * ldb;
  const __bf16* gB2 = gB0 + (long)128 * ldb;
  const __bf16* gB3 = gB0 + (long)192 * ldb;

  const unsigned ldsA0 = lds_off(&As[0][ar0 * LDW + ac0]);
  const unsigned ldsA1 = ldsA0 + 64u * LDW * 2u;
  const unsigned ldsB0 = lds_off(&Bs[0][ar0 * LDW + ac0]);
  const unsigned ldsB1 = ldsB0 + 64u * LDW * 2u;
  const unsigned ldsB2 = ldsB0 + 128u * LDW * 2u;
  const unsigned ldsB3 = ldsB0 + 192u * LDW * 2u;

  auto issue = [&](int kt, int buf) {
    const long ko = (long)kt * BK;
    const unsigned ao = (unsigned)buf * (unsigned)(BM * LDW * 2);
    const unsigned bo = (unsigned)buf * (unsigned)(BN * LDW * 2);
    ASYNC_B128(ldsA0 + ao, gA0 + ko);
    ASYNC_B128(ldsA1 + ao, gA1 + ko);
    ASYNC_B128(ldsB0 + bo, gB0 + ko);
    ASYNC_B128(ldsB1 + bo, gB1 + ko);
    ASYNC_B128(ldsB2 + bo, gB2 + ko);
    ASYNC_B128(ldsB3 + bo, gB3 + ko);
  };

  v8f acc[4][4];
#pragma unroll
  for (int i = 0; i < 4; ++i)
#pragma unroll
    for (int j = 0; j < 4; ++j)
#pragma unroll
      for (int r = 0; r < 8; ++r) acc[i][j][r] = 0.0f;

  issue(0, 0);
  wait_async0();
  __syncthreads();

  const int nkt = K / BK;
  for (int kt = 0; kt < nkt; ++kt) {
    const int cur = kt & 1;
    const int nxt = cur ^ 1;
    const bool more = (kt + 1) < nkt;
    if (more) issue(kt + 1, nxt);  // DMA next tile while computing this one

    // --- fragments from LDS (ISA 7.12.2 layouts) ---
    FragAB af[4];
#pragma unroll
    for (int i = 0; i < 4; ++i) {
      const __bf16* p = &As[cur][(wm * 64 + i * 16 + l16) * LDW + hf * 8];
      af[i].h[0] = *(const bfx8*)p;        // k = hf*8 .. +7
      af[i].h[1] = *(const bfx8*)(p + 16); // k = 16+hf*8 .. +7
    }
    FragAB bfr[4];
#pragma unroll
    for (int j = 0; j < 4; ++j) {
      const __bf16* p = &Bs[cur][(wn * 64 + j * 16 + l16) * LDW + hf * 8];
      bfr[j].h[0] = *(const bfx8*)p;       // k = hf*8 .. +7
      bfr[j].h[1] = *(const bfx8*)(p + 16); // k = 16+hf*8 .. +7
    }
#pragma unroll
    for (int i = 0; i < 4; ++i)
#pragma unroll
      for (int j = 0; j < 4; ++j)
        acc[i][j] = __builtin_amdgcn_wmma_f32_16x16x32_bf16(
            false, af[i].v, false, bfr[j].v, (short)0, acc[i][j], false, false);

    if (more) {
      wait_async0();     // own DMAs into nxt have landed in LDS
      __syncthreads();   // everyone's have; also fences reads of cur
    }
  }

  // --- epilogue: D layout m = r + 8*hf (+16*i +64*wm), n = l16 (+16*j +64*wn)
#pragma unroll
  for (int i = 0; i < 4; ++i)
#pragma unroll
    for (int j = 0; j < 4; ++j)
#pragma unroll
      for (int r = 0; r < 8; ++r) {
        long m = m0 + wm * 64 + i * 16 + hf * 8 + r;
        long n = n0 + wn * 64 + j * 16 + l16;
        float v = acc[i][j][r];
        if (OUT_BF16) {
          __bf16* C = (__bf16*)Cv + (long)bz * sC;
          C[OUT_TRANS ? n * ldc + m : m * ldc + n] = (__bf16)v;
        } else {
          float* C = (float*)Cv + (long)bz * sC;
          C[OUT_TRANS ? n * ldc + m : m * ldc + n] = v;
        }
      }
}

// ---------------------------------------------------------------------------
// Row softmax: one block per row of S [rows = B*T, width = T], writes bf16 P.
// ---------------------------------------------------------------------------
__global__ __launch_bounds__(256) void k_softmax(const float* __restrict__ S,
                                                 __bf16* __restrict__ P) {
  __shared__ float red[256];
  const long row = blockIdx.x;
  const float* s = S + row * (long)Tt;
  __bf16* p = P + row * (long)Tt;
  const int tid = threadIdx.x;

  float vals[8];
  float mx = -3.0e38f;
#pragma unroll
  for (int i = 0; i < 8; ++i) {
    vals[i] = s[tid + i * 256];
    mx = fmaxf(mx, vals[i]);
  }
  red[tid] = mx;
  __syncthreads();
  for (int o = 128; o > 0; o >>= 1) {
    if (tid < o) red[tid] = fmaxf(red[tid], red[tid + o]);
    __syncthreads();
  }
  mx = red[0];
  __syncthreads();

  float sum = 0.0f;
#pragma unroll
  for (int i = 0; i < 8; ++i) {
    vals[i] = __expf(vals[i] - mx);
    sum += vals[i];
  }
  red[tid] = sum;
  __syncthreads();
  for (int o = 128; o > 0; o >>= 1) {
    if (tid < o) red[tid] += red[tid + o];
    __syncthreads();
  }
  const float inv = 1.0f / red[0];
#pragma unroll
  for (int i = 0; i < 8; ++i) p[tid + i * 256] = (__bf16)(vals[i] * inv);
}

// ---------------------------------------------------------------------------
// Host-side orchestration
// ---------------------------------------------------------------------------
extern "C" void kernel_launch(void* const* d_in, const int* in_sizes, int n_in,
                              void* d_out, int out_size, void* d_ws, size_t ws_size,
                              hipStream_t stream) {
  const float* X  = (const float*)d_in[0];  // [B, T, D]
  const float* Wq = (const float*)d_in[1];  // [D, N]
  const float* Wk = (const float*)d_in[2];
  const float* Wv = (const float*)d_in[3];
  float* out = (float*)d_out;               // [B, T, N] f32

  char* ws = (char*)d_ws;
  __bf16* Xb  = (__bf16*)ws; ws += (size_t)Bb * Tt * Dd * 2;  // 32 MB
  __bf16* Wqt = (__bf16*)ws; ws += (size_t)Nd * Dd * 2;       //  2 MB
  __bf16* Wkt = (__bf16*)ws; ws += (size_t)Nd * Dd * 2;
  __bf16* Wvt = (__bf16*)ws; ws += (size_t)Nd * Dd * 2;
  __bf16* Qb  = (__bf16*)ws; ws += (size_t)Bb * Tt * Nd * 2;  // 32 MB
  __bf16* Kb  = (__bf16*)ws; ws += (size_t)Bb * Tt * Nd * 2;  // 32 MB
  __bf16* Vt  = (__bf16*)ws; ws += (size_t)Nd * Bb * Tt * 2;  // 32 MB, [N][B*T]
  float*  Sc  = (float*)ws;  ws += (size_t)Bb * Tt * Tt * 4;  // 128 MB
  __bf16* P   = (__bf16*)ws; ws += (size_t)Bb * Tt * Tt * 2;  // 64 MB

  // 1) precision conversion (bf16 keeps f32 exponent; right WMMA input type)
  {
    long nX = (long)Bb * Tt * Dd;
    k_cvt_bf16<<<(int)(nX / 4 / 256), 256, 0, stream>>>(X, Xb);
    dim3 gw(Nd / 32, Dd / 32);
    k_cvt_wt<<<gw, 256, 0, stream>>>(Wq, Wqt);
    k_cvt_wt<<<gw, 256, 0, stream>>>(Wk, Wkt);
    k_cvt_wt<<<gw, 256, 0, stream>>>(Wv, Wvt);
  }

  const dim3 blk(256);

  // 2) QKV projections: (B*T,1024) x (1024,1024). V written transposed so the
  //    P@V GEMM gets a K-contiguous B operand.
  {
    dim3 g(Nd / 256, (Bb * Tt) / 128, 1);
    k_gemm<true, false><<<g, blk, 0, stream>>>(Xb, Wqt, Qb, Dd, Dd, Dd, Nd, 0, 0, 0);
    k_gemm<true, false><<<g, blk, 0, stream>>>(Xb, Wkt, Kb, Dd, Dd, Dd, Nd, 0, 0, 0);
    k_gemm<true, true ><<<g, blk, 0, stream>>>(Xb, Wvt, Vt, Dd, Dd, Dd, Bb * Tt, 0, 0, 0);
  }

  // 3) scores[b] = Q[b] (2048x1024) x K[b]^T  -> f32 (K already K-contiguous)
  {
    dim3 g(Tt / 256, Tt / 128, Bb);
    k_gemm<false, false><<<g, blk, 0, stream>>>(
        Qb, Kb, Sc, Nd, Nd, Nd, Tt,
        (long)Tt * Nd, (long)Tt * Nd, (long)Tt * Tt);
  }

  // 4) softmax rows -> bf16 attention weights
  k_softmax<<<Bb * Tt, 256, 0, stream>>>(Sc, P);

  // 5) out[b] = P[b] (2048x2048) x V[b] (2048x1024), V supplied as Vt[n][b*T+k]
  {
    dim3 g(Nd / 256, Tt / 128, Bb);
    k_gemm<false, false><<<g, blk, 0, stream>>>(
        P, Vt, out, Tt, Tt, Bb * Tt, Nd,
        (long)Tt * Tt, (long)Tt, (long)Tt * Nd);
  }
  (void)in_sizes; (void)n_in; (void)out_size; (void)ws_size;
}
